// Attention_36344013258878
// MI455X (gfx1250) — matrix operand order repaired and checked by reading, not compiled, not run on hardware
//
#include <hip/hip_runtime.h>
#include <hip/hip_fp16.h>

typedef _Float16 v16h __attribute__((ext_vector_type(16)));
typedef _Float16 v8h  __attribute__((ext_vector_type(8)));
typedef float    v8f  __attribute__((ext_vector_type(8)));

#define WMMA16(A,B,C) \
  __builtin_amdgcn_wmma_f32_16x16x32_f16(false,(A),false,(B),(short)0,(C),false,false)

// Problem constants: B=4, N=2048, C=1024, H=16, HD=64, scale = 0.125

// ---------------------------------------------------------------------------
__global__ void cvt_f32_to_f16(const float* __restrict__ src,
                               _Float16* __restrict__ dst, int n) {
  int i = blockIdx.x * blockDim.x + threadIdx.x;
  if (i < n) dst[i] = (_Float16)src[i];
}

// ---------------------------------------------------------------------------
// Pack row-major f32 W[K,Nc] into WMMA B-matrix fragments:
// flat = ((nt*KT + kt)*32 + lane)*16 + i ; b[i] = W[kt*32+(lane>>4)*8+(i&7)+(i>>3)*16][nt*16+(lane&15)]
__global__ void pack_w_bfrag(const float* __restrict__ W,
                             _Float16* __restrict__ out, int K, int Nc) {
  int tid = blockIdx.x * blockDim.x + threadIdx.x;
  if (tid >= K * Nc) return;
  int i    = tid & 15;
  int lane = (tid >> 4) & 31;
  int rest = tid >> 9;
  int KT   = K >> 5;
  int kt   = rest % KT;
  int nt   = rest / KT;
  int k = kt * 32 + ((lane >> 4) << 3) + (i & 7) + ((i >> 3) << 4);
  int n = nt * 16 + (lane & 15);
  out[tid] = (_Float16)W[(size_t)k * Nc + n];
}

// ---------------------------------------------------------------------------
// A-fragment load helper: row-major f16 matrix, row = lane&15,
// halves at kbase=(lane>>4)*8 and kbase+16 (ISA 16-bit A layout).
__device__ __forceinline__ v16h load_afrag(const _Float16* p) {
  v8h lo = *(const v8h*)p;
  v8h hi = *(const v8h*)(p + 16);
  return __builtin_shufflevector(lo, hi, 0,1,2,3,4,5,6,7,8,9,10,11,12,13,14,15);
}

// Scatter one 16x16 f32 D-tile of the QKV GEMM into fragment-layout Q/K/V.
__device__ __forceinline__ void qkv_scatter(int mt, int nt, int lane, v8f c,
                                            _Float16* __restrict__ qpack,
                                            _Float16* __restrict__ kpack,
                                            _Float16* __restrict__ vpack) {
  const int j    = nt * 16 + (lane & 15);
  const int part = j >> 10;     // wave-uniform
  const int jc   = j & 1023;
  const int h    = jc >> 6;     // wave-uniform
  const int d    = jc & 63;
  const int mofs = (lane >> 4) << 3;
#pragma unroll
  for (int v = 0; v < 8; ++v) {
    int t  = mt * 16 + v + mofs;
    int b_ = t >> 11;
    int n  = t & 2047;
    int bh = b_ * 16 + h;
    float val = c[v];
    if (part == 0) {            // Q -> B-frag of Q^T, scale folded in
      int qt = n >> 4, ktd = d >> 5, dd = d & 31;
      int lp = (((dd >> 3) & 1) << 4) | (n & 15);
      int ii = (dd & 7) | ((dd >> 4) << 3);
      size_t idx = ((((size_t)bh * 128 + qt) * 2 + ktd) * 32 + lp) * 16 + ii;
      qpack[idx] = (_Float16)(val * 0.125f);
    } else if (part == 1) {     // K -> A-frag (16 keys x 32 d)
      int ktile = n >> 4, ktd = d >> 5, dd = d & 31;
      int lp = (((dd >> 3) & 1) << 4) | (n & 15);
      int ii = (dd & 7) | ((dd >> 4) << 3);
      size_t idx = ((((size_t)bh * 128 + ktile) * 2 + ktd) * 32 + lp) * 16 + ii;
      kpack[idx] = (_Float16)val;
    } else {                    // V -> A-frag of V^T (16 d x 32 keys)
      int ktv = n >> 5, ntd = d >> 4, ko = n & 31;
      int lp = ((ko >> 4) << 4) | (d & 15);
      int ii = ko & 15;
      size_t idx = ((((size_t)bh * 64 + ktv) * 4 + ntd) * 32 + lp) * 16 + ii;
      vpack[idx] = (_Float16)val;
    }
  }
}

// ---------------------------------------------------------------------------
// QKV GEMM: [8192,1024] x [1024,3072], 64x64 output per wave (4x4 subtiles):
// 8 fragment loads -> 16 WMMAs per K-step.
__global__ __launch_bounds__(256, 1) void qkv_gemm(
    const _Float16* __restrict__ xh, const _Float16* __restrict__ wpk,
    _Float16* __restrict__ qpack, _Float16* __restrict__ kpack,
    _Float16* __restrict__ vpack) {
  const int lane = threadIdx.x & 31;
  const int wid  = blockIdx.x * 8 + (threadIdx.x >> 5);
  const int NB = 48;                        // 3072/64
  const int mb = wid / NB, nb = wid % NB;   // exact: 128*48 = 6144 waves
  const int row = lane & 15;
  const int kb  = (lane >> 4) << 3;
  const v16h* bv = (const v16h*)wpk;

  v8f acc[16];
#pragma unroll
  for (int t = 0; t < 16; ++t) acc[t] = (v8f){0.f,0.f,0.f,0.f,0.f,0.f,0.f,0.f};

#pragma unroll 2
  for (int kt = 0; kt < 32; ++kt) {
    v16h a[4], b[4];
#pragma unroll
    for (int s = 0; s < 4; ++s) {
      a[s] = load_afrag(xh + (size_t)(mb * 64 + s * 16 + row) * 1024 + kt * 32 + kb);
      b[s] = bv[((size_t)(nb * 4 + s) * 32 + kt) * 32 + lane];
    }
#pragma unroll
    for (int i = 0; i < 4; ++i)
#pragma unroll
      for (int jn = 0; jn < 4; ++jn)
        acc[i * 4 + jn] = WMMA16(a[i], b[jn], acc[i * 4 + jn]);
  }

#pragma unroll
  for (int i = 0; i < 4; ++i)
#pragma unroll
    for (int jn = 0; jn < 4; ++jn)
      qkv_scatter(mb * 4 + i, nb * 4 + jn, lane, acc[i * 4 + jn],
                  qpack, kpack, vpack);
}

// ---------------------------------------------------------------------------
// Flash-attention, transposed (S^T = K*Q^T). One wave handles 64 queries
// (4 query tiles) so each K/V fragment feeds 4x the WMMAs.
__global__ __launch_bounds__(256, 1) void attention(
    const _Float16* __restrict__ qpack, const _Float16* __restrict__ kpack,
    const _Float16* __restrict__ vpack, _Float16* __restrict__ attnh) {
  const int lane = threadIdx.x & 31;
  const int wid  = blockIdx.x * 8 + (threadIdx.x >> 5);
  const int bh = wid >> 5;   // b*16 + h   (2048 waves total)
  const int qg = wid & 31;   // group of 4 query tiles

  const v16h* qv = (const v16h*)qpack;
  const v16h* kv = (const v16h*)kpack;
  const v16h* vv = (const v16h*)vpack;

  v16h qf[4][2];
#pragma unroll
  for (int qi = 0; qi < 4; ++qi) {
    size_t qb = (((size_t)bh * 128 + qg * 4 + qi) * 2) * 32 + lane;
    qf[qi][0] = qv[qb];
    qf[qi][1] = qv[qb + 32];
  }

  float mrun[4], lrun[4];
  v8f acc[4][4];
#pragma unroll
  for (int qi = 0; qi < 4; ++qi) {
    mrun[qi] = -3.0e38f;
    lrun[qi] = 0.f;
#pragma unroll
    for (int t = 0; t < 4; ++t) acc[qi][t] = (v8f){0.f,0.f,0.f,0.f,0.f,0.f,0.f,0.f};
  }
  const v8f zero = {0.f,0.f,0.f,0.f,0.f,0.f,0.f,0.f};
  const bool lo_half = lane < 16;

#pragma unroll 1
  for (int kt = 0; kt < 64; ++kt) {          // 32 keys per chunk
    size_t kbi = ((size_t)bh * 128 + (size_t)kt * 2) * 2 * 32 + lane;
    size_t vbi = ((size_t)bh * 64 + kt) * 4 * 32 + lane;
    if (kt < 63) {                           // prefetch next chunk (L2-resident)
      __builtin_prefetch((const void*)(kv + kbi + 128), 0, 1);
      __builtin_prefetch((const void*)(vv + vbi + 128), 0, 1);
    }
    v16h kf0 = kv[kbi],      kf1 = kv[kbi + 32];
    v16h kf2 = kv[kbi + 64], kf3 = kv[kbi + 96];
    v16h vf0 = vv[vbi],      vf1 = vv[vbi + 32];
    v16h vf2 = vv[vbi + 64], vf3 = vv[vbi + 96];

#pragma unroll
    for (int qi = 0; qi < 4; ++qi) {
      v8f s0 = WMMA16(kf0, qf[qi][0], zero);
      s0     = WMMA16(kf1, qf[qi][1], s0);
      v8f s1 = WMMA16(kf2, qf[qi][0], zero);
      s1     = WMMA16(kf3, qf[qi][1], s1);

      float mx = s0[0];
#pragma unroll
      for (int v = 0; v < 8; ++v) { mx = fmaxf(mx, s0[v]); mx = fmaxf(mx, s1[v]); }
      mx = fmaxf(mx, __shfl_xor(mx, 16, 32));
      float mnew  = fmaxf(mrun[qi], mx);
      float alpha = __expf(mrun[qi] - mnew);

      float p0[8], p1[8], psum = 0.f;
#pragma unroll
      for (int v = 0; v < 8; ++v) {
        p0[v] = __expf(s0[v] - mnew);
        p1[v] = __expf(s1[v] - mnew);
        psum += p0[v] + p1[v];
      }
      lrun[qi] = lrun[qi] * alpha + psum;
      mrun[qi] = mnew;
#pragma unroll
      for (int t = 0; t < 4; ++t)
#pragma unroll
        for (int v = 0; v < 8; ++v) acc[qi][t][v] *= alpha;

      // P^T B-fragment (32 keys x 16 queries) via half-wave swaps
      v16h pf;
#pragma unroll
      for (int v = 0; v < 8; ++v) {
        float o0 = __shfl_xor(p0[v], 16, 32);
        float o1 = __shfl_xor(p1[v], 16, 32);
        pf[v]     = (_Float16)(lo_half ? p0[v] : o1);
        pf[v + 8] = (_Float16)(lo_half ? o0    : p1[v]);
      }

      acc[qi][0] = WMMA16(vf0, pf, acc[qi][0]);
      acc[qi][1] = WMMA16(vf1, pf, acc[qi][1]);
      acc[qi][2] = WMMA16(vf2, pf, acc[qi][2]);
      acc[qi][3] = WMMA16(vf3, pf, acc[qi][3]);
    }
  }

  const int b_ = bh >> 4, h = bh & 15;
  const int dof = (lane >> 4) << 3;
#pragma unroll
  for (int qi = 0; qi < 4; ++qi) {
    float ltot = lrun[qi] + __shfl_xor(lrun[qi], 16, 32);
    float rinv = 1.0f / ltot;
    int q = (qg * 4 + qi) * 16 + (lane & 15);
    _Float16* orow = attnh + ((size_t)b_ * 2048 + q) * 1024 + h * 64;
#pragma unroll
    for (int t = 0; t < 4; ++t)
#pragma unroll
      for (int v = 0; v < 8; ++v)
        orow[t * 16 + v + dof] = (_Float16)(acc[qi][t][v] * rinv);
  }
}

// ---------------------------------------------------------------------------
// Output projection: [8192,1024] x [1024,1024] + bias, 64x64 per wave.
__global__ __launch_bounds__(256, 1) void proj_gemm(
    const _Float16* __restrict__ ah, const _Float16* __restrict__ wpk,
    const float* __restrict__ bias, float* __restrict__ out) {
  const int lane = threadIdx.x & 31;
  const int wid  = blockIdx.x * 8 + (threadIdx.x >> 5);
  const int NB = 16;                        // 1024/64
  const int mb = wid / NB, nb = wid % NB;   // exact: 128*16 = 2048 waves
  const int row = lane & 15;
  const int kb  = (lane >> 4) << 3;
  const v16h* bv = (const v16h*)wpk;

  v8f acc[16];
#pragma unroll
  for (int t = 0; t < 16; ++t) acc[t] = (v8f){0.f,0.f,0.f,0.f,0.f,0.f,0.f,0.f};

#pragma unroll 2
  for (int kt = 0; kt < 32; ++kt) {
    v16h a[4], b[4];
#pragma unroll
    for (int s = 0; s < 4; ++s) {
      a[s] = load_afrag(ah + (size_t)(mb * 64 + s * 16 + row) * 1024 + kt * 32 + kb);
      b[s] = bv[((size_t)(nb * 4 + s) * 32 + kt) * 32 + lane];
    }
#pragma unroll
    for (int i = 0; i < 4; ++i)
#pragma unroll
      for (int jn = 0; jn < 4; ++jn)
        acc[i * 4 + jn] = WMMA16(a[i], b[jn], acc[i * 4 + jn]);
  }

  const int mofs = (lane >> 4) << 3;
#pragma unroll
  for (int i = 0; i < 4; ++i) {
#pragma unroll
    for (int jn = 0; jn < 4; ++jn) {
      int j = (nb * 4 + jn) * 16 + (lane & 15);
      float bj = bias[j];
      v8f c = acc[i * 4 + jn];
#pragma unroll
      for (int v = 0; v < 8; ++v)
        out[(size_t)((mb * 4 + i) * 16 + v + mofs) * 1024 + j] = c[v] + bj;
    }
  }
}

// ---------------------------------------------------------------------------
extern "C" void kernel_launch(void* const* d_in, const int* in_sizes, int n_in,
                              void* d_out, int out_size, void* d_ws, size_t ws_size,
                              hipStream_t stream) {
  const float* x     = (const float*)d_in[0];  // [4,2048,1024]
  const float* wqkv  = (const float*)d_in[1];  // [1024,3072]
  const float* wproj = (const float*)d_in[2];  // [1024,1024]
  const float* bproj = (const float*)d_in[3];  // [1024]
  float* out = (float*)d_out;                  // [4,2048,1024]

  char* ws = (char*)d_ws;
  _Float16* xh     = (_Float16*)ws; ws += (size_t)8192 * 1024 * 2;   // 16 MB
  _Float16* wqkvp  = (_Float16*)ws; ws += (size_t)1024 * 3072 * 2;   //  6 MB
  _Float16* wprojp = (_Float16*)ws; ws += (size_t)1024 * 1024 * 2;   //  2 MB
  _Float16* qp     = (_Float16*)ws; ws += (size_t)8388608 * 2;       // 16 MB
  _Float16* kp     = (_Float16*)ws; ws += (size_t)8388608 * 2;       // 16 MB
  _Float16* vp     = (_Float16*)ws; ws += (size_t)8388608 * 2;       // 16 MB
  _Float16* ah     = (_Float16*)ws; ws += (size_t)8192 * 1024 * 2;   // 16 MB

  cvt_f32_to_f16<<<32768, 256, 0, stream>>>(x, xh, 8192 * 1024);
  pack_w_bfrag<<<12288, 256, 0, stream>>>(wqkv, wqkvp, 1024, 3072);
  pack_w_bfrag<<<4096, 256, 0, stream>>>(wproj, wprojp, 1024, 1024);
  qkv_gemm<<<768, 256, 0, stream>>>(xh, wqkvp, qp, kp, vp);       // 128*48 waves
  attention<<<256, 256, 0, stream>>>(qp, kp, vp, ah);             // 64*32 waves
  proj_gemm<<<256, 256, 0, stream>>>(ah, wprojp, bproj, out);     // 128*16 waves
}
